// Aggregation1_41979010351806
// MI455X (gfx1250) — matrix operand order repaired and checked by reading, not compiled, not run on hardware
//
#include <hip/hip_runtime.h>
#include <hip/hip_bf16.h>

typedef __attribute__((ext_vector_type(16))) _Float16 v16h;
typedef __attribute__((ext_vector_type(8)))  float    v8f;

#define PS   5
#define PADP 4      // 2*(PS//2)
#define DIL  2
#define CCH  3
#define H0   384
#define HH   392    // H0 + 2*PADP
#define TILES ((HH + 15) / 16)   // 25

// ---------------------------------------------------------------------------
// Kernel 1: zero the accumulator [npix][4] = {ch0, ch1, ch2, weight}
// ---------------------------------------------------------------------------
__global__ void zero_kernel(float* __restrict__ p, long long n) {
  long long i = (long long)blockIdx.x * blockDim.x + threadIdx.x;
  if (i < n) p[i] = 0.0f;
}

// ---------------------------------------------------------------------------
// Kernel 2: scatter-add. One thread per (patch n, position pos in 5x5).
// acc layout interleaved: acc[lin*4 + {0,1,2}] = channel sums, +3 = weight.
// ---------------------------------------------------------------------------
__global__ void scatter_kernel(const float* __restrict__ x,
                               const int* __restrict__ nlInds,
                               float* __restrict__ acc,
                               long long M, long long npix) {
  long long id = (long long)blockIdx.x * blockDim.x + threadIdx.x;
  if (id >= M) return;
  long long n = id / 25;
  int pos = (int)(id - n * 25);
  int pi = pos / PS, pj = pos - pi * PS;

  int ti = nlInds[n * 3 + 0];
  int hi = nlInds[n * 3 + 1] + PADP + DIL * pi;
  int wi = nlInds[n * 3 + 2] + PADP + DIL * pj;
  long long lin = ((long long)ti * HH + hi) * HH + wi;
  if (lin < 0 || lin >= npix) return;

  float v0 = x[n * 75 + 0 * 25 + pos];
  float v1 = x[n * 75 + 1 * 25 + pos];
  float v2 = x[n * 75 + 2 * 25 + pos];
  float* a = acc + lin * 4;
  atomicAdd(a + 0, v0);
  atomicAdd(a + 1, v1);
  atomicAdd(a + 2, v2);
  atomicAdd(a + 3, 1.0f);
}

// ---------------------------------------------------------------------------
// Kernel 3: normalize + separable 3x3 binomial blur via WMMA.
// One wave (32 threads) per 16x16 output tile per (t, channel).
//   T(16x18)  = Wv(16x32 band, f16) x X(18x18 halo, f16)   [2 WMMAs: cols 0..15, 16..17]
//   out(16x16)= T(16x32 pad, f16)   x Wh(32x16 band, f16)  [1 WMMA]
// X values are acc[ch]/acc[weight] with reflect(1) boundary handling.
// Output written in "bflat" layout: bflat[((t*H+row)*W+col)*3 + c].
// ---------------------------------------------------------------------------
__global__ __launch_bounds__(32)
void blur_wmma_kernel(const float* __restrict__ acc, float* __restrict__ bflat) {
  const int tile = blockIdx.x;
  const int tx = tile % TILES, ty = tile / TILES;
  const int c  = blockIdx.y;
  const int tt = blockIdx.z;
  const int r0 = ty * 16, c0 = tx * 16;
  const int lane = threadIdx.x;

  __shared__ float xs[18][20];    // halo tile (padded stride)
  __shared__ float tmp[16][20];   // vertical-blur intermediate, 16x18 valid

  // Cooperative halo load with reflect(1) at image borders; out-of-tile-range
  // rows/cols beyond reflect stay in-bounds (garbage there is clipped later).
  const float* accT = acc + (long long)tt * HH * HH * 4;
  for (int i = lane; i < 18 * 18; i += 32) {
    int k = i / 18, nn = i - k * 18;
    int rr = r0 - 1 + k;
    int cc = c0 - 1 + nn;
    rr = (rr < 0) ? -rr : rr;  rr = (rr >= HH) ? (2 * HH - 2 - rr) : rr;
    cc = (cc < 0) ? -cc : cc;  cc = (cc >= HH) ? (2 * HH - 2 - cc) : cc;
    rr = (rr < 0) ? 0 : (rr > HH - 1 ? HH - 1 : rr);
    cc = (cc < 0) ? 0 : (cc > HH - 1 ? HH - 1 : cc);
    long long p = (long long)rr * HH + cc;
    float w = accT[p * 4 + 3];
    xs[k][nn] = accT[p * 4 + c] / w;
  }
  __syncthreads();

  // Per-lane operand packing (ISA 7.12.2, wave32):
  //  A (16x32 f16): lanes 0-15 -> M=lane, K in {0..7,16..23}; lanes 16-31 -> M=lane-16, K in {8..15,24..31}
  //  B (32x16 f16): lanes 0-15 -> K in {0..7,16..23}; lanes 16-31 -> K in {8..15,24..31}; N = lane&15
  const int Mrow  = lane & 15;
  const int kbase = (lane < 16) ? 0 : 8;    // A-matrix K base
  const int kb    = (lane < 16) ? 0 : 8;    // B-matrix K base
  const int Ncol  = lane & 15;

  v16h a1, b1lo, b1hi;
#pragma unroll
  for (int v = 0; v < 8; ++v) {
#pragma unroll
    for (int j = 0; j < 2; ++j) {
      int Ka = kbase + ((v < 4) ? 2 * v : 16 + 2 * (v - 4)) + j;
      // Wv[M][K]: vblur row M uses X rows M, M+1, M+2 (halo-offset indexing)
      float wv = (Ka == Mrow) ? 0.25f : (Ka == Mrow + 1) ? 0.5f
               : (Ka == Mrow + 2) ? 0.25f : 0.0f;
      a1[2 * v + j] = (_Float16)wv;

      int Kb = kb + ((v < 4) ? 2 * v : 16 + 2 * (v - 4)) + j;
      float xlo = (Kb < 18) ? xs[Kb][Ncol] : 0.0f;               // X cols 0..15
      float xhi = (Kb < 18 && Ncol < 2) ? xs[Kb][16 + Ncol] : 0.0f; // X cols 16..17
      b1lo[2 * v + j] = (_Float16)xlo;
      b1hi[2 * v + j] = (_Float16)xhi;
    }
  }

  v8f z = {};
  v8f t1 = __builtin_amdgcn_wmma_f32_16x16x32_f16(false, a1, false, b1lo,
                                                  (short)0, z, false, false);
  v8f t2 = __builtin_amdgcn_wmma_f32_16x16x32_f16(false, a1, false, b1hi,
                                                  (short)0, z, false, false);

  // D layout -> LDS (row, col): lane gives N=lane&15, VGPR r gives M = r + 8*(lane>=16)
  {
    int mb = (lane >> 4) * 8;
#pragma unroll
    for (int r = 0; r < 8; ++r) {
      tmp[mb + r][Ncol] = t1[r];
      if (Ncol < 2) tmp[mb + r][16 + Ncol] = t2[r];
    }
  }
  __syncthreads();

  // Horizontal pass: A2 = tmp (16x18, padded to 16x32), B2 = Wh band.
  v16h a2, b2;
#pragma unroll
  for (int v = 0; v < 8; ++v) {
#pragma unroll
    for (int j = 0; j < 2; ++j) {
      int Ka = kbase + ((v < 4) ? 2 * v : 16 + 2 * (v - 4)) + j;
      float tv = (Ka < 18) ? tmp[Mrow][Ka] : 0.0f;
      a2[2 * v + j] = (_Float16)tv;

      int Kb = kb + ((v < 4) ? 2 * v : 16 + 2 * (v - 4)) + j;
      // Wh[K][N]: out col N uses tmp cols N, N+1, N+2 (halo-offset indexing)
      float wh = (Kb == Ncol) ? 0.25f : (Kb == Ncol + 1) ? 0.5f
               : (Kb == Ncol + 2) ? 0.25f : 0.0f;
      b2[2 * v + j] = (_Float16)wh;
    }
  }

  v8f outd = __builtin_amdgcn_wmma_f32_16x16x32_f16(false, a2, false, b2,
                                                    (short)0, z, false, false);

  // Store D, clipping the partial edge tiles (25*16 = 400 > 392).
  int col = c0 + Ncol;
  if (col < HH) {
    int mb = (lane >> 4) * 8;
#pragma unroll
    for (int r = 0; r < 8; ++r) {
      int row = r0 + mb + r;
      if (row < HH) {
        long long p = ((long long)tt * HH + row) * HH + col;
        bflat[p * 3 + c] = outd[r];
      }
    }
  }
}

// ---------------------------------------------------------------------------
// Kernel 4: gather. One thread per (patch n, position pos). Reads 12B
// contiguous from L2-resident bflat, writes coalesced 177MB output stream.
// ---------------------------------------------------------------------------
__global__ void gather_kernel(const float* __restrict__ bflat,
                              const int* __restrict__ nlInds,
                              float* __restrict__ out,
                              long long M, long long npix) {
  long long id = (long long)blockIdx.x * blockDim.x + threadIdx.x;
  if (id >= M) return;
  long long n = id / 25;
  int pos = (int)(id - n * 25);
  int pi = pos / PS, pj = pos - pi * PS;

  int ti = nlInds[n * 3 + 0];
  int hi = nlInds[n * 3 + 1] + PADP + DIL * pi;
  int wi = nlInds[n * 3 + 2] + PADP + DIL * pj;
  long long lin = ((long long)ti * HH + hi) * HH + wi;
  lin = (lin < 0) ? 0 : (lin >= npix ? npix - 1 : lin);

  float b0 = bflat[lin * 3 + 0];
  float b1 = bflat[lin * 3 + 1];
  float b2 = bflat[lin * 3 + 2];
  out[n * 75 + 0 * 25 + pos] = b0;
  out[n * 75 + 1 * 25 + pos] = b1;
  out[n * 75 + 2 * 25 + pos] = b2;
}

// ---------------------------------------------------------------------------
extern "C" void kernel_launch(void* const* d_in, const int* in_sizes, int n_in,
                              void* d_out, int out_size, void* d_ws, size_t ws_size,
                              hipStream_t stream) {
  const float* x      = (const float*)d_in[0];
  // d_in[1] = nlDists (unused by the reference computation)
  const int*   nlInds = (const int*)d_in[2];
  float*       out    = (float*)d_out;

  const long long N = (long long)in_sizes[0] / 75;       // t * p patches
  const int       T = (int)(N / ((long long)H0 * H0));   // t
  const long long npix = (long long)T * HH * HH;         // t*H*W
  const long long M = N * 25;                            // patch-positions

  float* acc   = (float*)d_ws;          // [npix][4] : ch0 ch1 ch2 weight
  float* bflat = acc + npix * 4;        // [npix][3] : blurred image, bflat layout

  {
    long long zn = npix * 4;
    int blocks = (int)((zn + 255) / 256);
    zero_kernel<<<blocks, 256, 0, stream>>>(acc, zn);
  }
  {
    int blocks = (int)((M + 255) / 256);
    scatter_kernel<<<blocks, 256, 0, stream>>>(x, nlInds, acc, M, npix);
  }
  {
    dim3 grid(TILES * TILES, CCH, T);
    blur_wmma_kernel<<<grid, 32, 0, stream>>>(acc, bflat);
  }
  {
    int blocks = (int)((M + 255) / 256);
    gather_kernel<<<blocks, 256, 0, stream>>>(bflat, nlInds, out, M, npix);
  }
}
